// BaseEncoder_71536975283029
// MI455X (gfx1250) — matrix operand order repaired and checked by reading, not compiled, not run on hardware
//
#include <hip/hip_runtime.h>
#include <hip/hip_bf16.h>
#include <math.h>

typedef __attribute__((ext_vector_type(16))) __bf16 v16bf;
typedef __attribute__((ext_vector_type(8)))  float  v8f;
typedef __attribute__((ext_vector_type(4)))  unsigned int u32x4;
typedef __attribute__((ext_vector_type(8)))  int i32x8;
typedef __attribute__((ext_vector_type(4)))  int i32x4;

namespace cfg {
constexpr int B = 4, N = 2048, D = 512, H = 8, K = 9, P = 8, E = 4, HID = 2048;
constexpr int M = B * N;        // 8192 tokens
constexpr int HDIM = D / H;     // 64
constexpr int HALF = HDIM / 2;  // 32 == wave32 lanes
}

#if __has_builtin(__builtin_amdgcn_tensor_load_to_lds)
#define USE_TDM 1
#else
#define USE_TDM 0
#endif

// ---------------------------------------------------------------- utilities

__device__ inline float gelu_f(float x) {
  return 0.5f * x * (1.0f + erff(x * 0.7071067811865476f));
}

__device__ inline float wave_sum(float v) {
#pragma unroll
  for (int m = 16; m > 0; m >>= 1) v += __shfl_xor(v, m, 32);
  return v;
}

__device__ inline float block_sum(float v) {
  __shared__ float sh[8];
  __shared__ float tot;
  const int lane = threadIdx.x & 31, w = threadIdx.x >> 5;
  v = wave_sum(v);
  if (lane == 0) sh[w] = v;
  __syncthreads();
  if (threadIdx.x == 0) {
    float t = 0.f;
    for (int i = 0; i < (int)(blockDim.x >> 5); ++i) t += sh[i];
    tot = t;
  }
  __syncthreads();
  return tot;
}

#if USE_TDM
// LDS byte offset of a pointer into a __shared__ array (ptrtoint of AS(3)).
__device__ inline unsigned lds_offset_of(const void* p) {
  return (unsigned)(unsigned long long)(__attribute__((address_space(3))) const void*)p;
}

// Issue a TDM 2D tile load: tile_dim0 = `cols` contiguous bf16 elements along
// the fastest dim, tile_dim1 = `rows` lines with stride `stride_elems`.
// D# layout per CDNA5 ISA 8.3/8.4 (count=1, type=2="image", data_size=1 -> 2B).
__device__ inline void tdm_load_2d(const void* gaddr, unsigned lds_off,
                                   unsigned rows, unsigned cols,
                                   unsigned tensor_rows, unsigned tensor_cols,
                                   unsigned stride_elems) {
  const unsigned long long ga = (unsigned long long)gaddr;
  u32x4 g0;
  g0[0] = 1u;                                   // count=1, user descriptor
  g0[1] = lds_off;                              // lds_addr (bytes)
  g0[2] = (unsigned)(ga & 0xffffffffu);         // global_addr[31:0]
  g0[3] = (unsigned)((ga >> 32) & 0x01ffffffu)  // global_addr[56:32]
          | (2u << 30);                         // type=2
  i32x8 g1;
  g1[0] = (int)(1u << 16);                      // data_size=1 (2 bytes)
  g1[1] = (int)((tensor_cols & 0xffffu) << 16);         // tensor_dim0[15:0]
  g1[2] = (int)((tensor_cols >> 16) |
                ((tensor_rows & 0xffffu) << 16));        // dim0 hi | dim1 lo
  g1[3] = (int)((tensor_rows >> 16) | (cols << 16));     // dim1 hi | tile_dim0
  g1[4] = (int)(rows & 0xffffu);                         // tile_dim1 (dim2=0)
  g1[5] = (int)stride_elems;                             // tensor_dim0_stride lo
  g1[6] = 0;                                             // stride hi | dim1_stride
  g1[7] = 0;
  i32x4 z4 = {0, 0, 0, 0};
#if __clang_major__ >= 23
  i32x8 z8 = {0, 0, 0, 0, 0, 0, 0, 0};
  __builtin_amdgcn_tensor_load_to_lds(g0, g1, z4, z4, z8, 0);
#else
  __builtin_amdgcn_tensor_load_to_lds(g0, g1, z4, z4, 0);
#endif
}
#endif  // USE_TDM

// ---------------------------------------------------------------- conversions

// convert + transpose: src[K,N] f32 -> dst[N,K] bf16 (weights, once per launch)
__global__ void f2bf_t_kernel(const float* __restrict__ src,
                              __bf16* __restrict__ dst, int Kd, int Nd) {
  const int i = blockIdx.x * blockDim.x + threadIdx.x;
  if (i >= Kd * Nd) return;
  const int n = i % Nd, k = i / Nd;  // coalesced read
  dst[(size_t)n * Kd + k] = (__bf16)src[i];
}

// ---------------------------------------------------------------- LayerNorm

__global__ __launch_bounds__(256) void ln_kernel(
    const float* __restrict__ in, const float* __restrict__ g,
    const float* __restrict__ b, float eps, float* __restrict__ outF,
    __bf16* __restrict__ outB, int Dd) {
  const int row = blockIdx.x;
  const float* x = in + (size_t)row * Dd;
  float s = 0.f;
  for (int d = threadIdx.x; d < Dd; d += blockDim.x) s += x[d];
  const float mean = block_sum(s) / (float)Dd;
  float v = 0.f;
  for (int d = threadIdx.x; d < Dd; d += blockDim.x) {
    const float t = x[d] - mean;
    v += t * t;
  }
  const float var = block_sum(v) / (float)Dd;
  const float inv = rsqrtf(var + eps);
  for (int d = threadIdx.x; d < Dd; d += blockDim.x) {
    const float y = (x[d] - mean) * inv * g[d] + b[d];
    if (outF) outF[(size_t)row * Dd + d] = y;
    if (outB) outB[(size_t)row * Dd + d] = (__bf16)y;
  }
}

// ---------------------------------------------------------------- WMMA GEMM
// C[M,N] = epilogue( A[M,K](bf16,row-major) @ WT[N,K](bf16, pre-transposed) )
// LDS tiles [128 rows][64 K], double-buffered; TDM streams the next K-tile
// (wave 0 issues tensor_load_to_lds, waits tensorcnt<=2 so the current tile is
// resident while the next pair is in flight). 16 WMMAs per staged tile.

__global__ __launch_bounds__(256) void gemm_bf16_wmma(
    const __bf16* __restrict__ A, const __bf16* __restrict__ WT,
    const float* __restrict__ bias, const float* __restrict__ residual,
    const float* __restrict__ rowscale, int rs_stride,
    float* __restrict__ outF, __bf16* __restrict__ outB,
    int M_, int N_, int K_, int act, int accumulate) {
  __shared__ __bf16 As[2][128][64];
  __shared__ __bf16 Bs[2][128][64];

  const int tid = threadIdx.x;
  const int lane = tid & 31;
  const int wave = tid >> 5;
  const int wm = wave & 3;   // 4 wave-rows * 32
  const int wn = wave >> 2;  // 2 wave-cols * 64
  const int bm = blockIdx.y * 128;
  const int bn = blockIdx.x * 128;

  v8f acc[2][4] = {};
  const int T = K_ >> 6;  // K tiles of 64 (all K here are multiples of 64)

#if USE_TDM
  if (wave == 0) {
    tdm_load_2d(A + (size_t)bm * K_, lds_offset_of(&As[0][0][0]), 128u, 64u,
                (unsigned)(M_ - bm), (unsigned)K_, (unsigned)K_);
    tdm_load_2d(WT + (size_t)bn * K_, lds_offset_of(&Bs[0][0][0]), 128u, 64u,
                (unsigned)(N_ - bn), (unsigned)K_, (unsigned)K_);
  }
#endif

  for (int t = 0; t < T; ++t) {
    const int buf = t & 1;
#if USE_TDM
    if (wave == 0) {
      if (t + 1 < T) {
        const int k1 = (t + 1) << 6;
        tdm_load_2d(A + (size_t)bm * K_ + k1,
                    lds_offset_of(&As[buf ^ 1][0][0]), 128u, 64u,
                    (unsigned)(M_ - bm), (unsigned)(K_ - k1), (unsigned)K_);
        tdm_load_2d(WT + (size_t)bn * K_ + k1,
                    lds_offset_of(&Bs[buf ^ 1][0][0]), 128u, 64u,
                    (unsigned)(N_ - bn), (unsigned)(K_ - k1), (unsigned)K_);
        // 4 ops outstanding; in-order retire => <=2 means tile t has landed.
        __builtin_amdgcn_s_wait_tensorcnt(2);
      } else {
        __builtin_amdgcn_s_wait_tensorcnt(0);
      }
    }
    __syncthreads();
#else
    {  // cooperative fallback: 32 contiguous bf16 per thread per matrix
      const int lr = tid >> 1;
      const int lc = (tid & 1) << 5;
      const int k0 = t << 6;
      uint4 av[4] = {}, bv[4] = {};
      if (bm + lr < M_) {
        const uint4* s = (const uint4*)(A + (size_t)(bm + lr) * K_ + k0 + lc);
#pragma unroll
        for (int u = 0; u < 4; ++u) av[u] = s[u];
      }
      if (bn + lr < N_) {
        const uint4* s = (const uint4*)(WT + (size_t)(bn + lr) * K_ + k0 + lc);
#pragma unroll
        for (int u = 0; u < 4; ++u) bv[u] = s[u];
      }
#pragma unroll
      for (int u = 0; u < 4; ++u) {
        *(uint4*)&As[buf][lr][lc + u * 8] = av[u];
        *(uint4*)&Bs[buf][lr][lc + u * 8] = bv[u];
      }
      __syncthreads();
    }
#endif

    // A fragment: lanes 0-15 rows, K chunks 0-7/16-23; lanes 16-31: 8-15/24-31.
    const int arow = lane & 15;
    const int kbA = (lane >= 16) ? 8 : 0;
    const int kbB = (lane >= 16) ? 8 : 0;
#pragma unroll
    for (int sub = 0; sub < 2; ++sub) {
      const int kb = sub << 5;
      v16bf afr[2];
#pragma unroll
      for (int i = 0; i < 2; ++i) {
        union { uint4 u[2]; v16bf v; } tt;
        tt.u[0] = *(const uint4*)&As[buf][wm * 32 + i * 16 + arow][kb + kbA];
        tt.u[1] = *(const uint4*)&As[buf][wm * 32 + i * 16 + arow][kb + kbA + 16];
        afr[i] = tt.v;
      }
      // B fragment (from WT tile): lanes 0-15 -> K 0-7/16-23, lanes 16-31 -> 8-15/24-31.
      v16bf bfr[4];
#pragma unroll
      for (int j = 0; j < 4; ++j) {
        const int c = wn * 64 + j * 16 + (lane & 15);
        union { uint4 u[2]; v16bf v; } tt;
        tt.u[0] = *(const uint4*)&Bs[buf][c][kb + kbB];
        tt.u[1] = *(const uint4*)&Bs[buf][c][kb + kbB + 16];
        bfr[j] = tt.v;
      }
#pragma unroll
      for (int i = 0; i < 2; ++i)
#pragma unroll
        for (int j = 0; j < 4; ++j)
          acc[i][j] = __builtin_amdgcn_wmma_f32_16x16x32_bf16(
              false, afr[i], false, bfr[j], (short)0, acc[i][j], false, false);
    }
    __syncthreads();
  }

  // epilogue: C/D layout -> lanes 0-15: N=lane, M=r; lanes 16-31: N=lane-16, M=8+r
#pragma unroll
  for (int i = 0; i < 2; ++i) {
#pragma unroll
    for (int j = 0; j < 4; ++j) {
      const int gn = bn + wn * 64 + j * 16 + (lane & 15);
      if (gn >= N_) continue;
      const float bia = bias ? bias[gn] : 0.0f;
#pragma unroll
      for (int r = 0; r < 8; ++r) {
        const int gm = bm + wm * 32 + i * 16 + ((lane >= 16) ? 8 : 0) + r;
        if (gm >= M_) continue;
        float v = acc[i][j][r] + bia;
        if (act == 1) v = gelu_f(v);
        if (rowscale) v *= rowscale[(size_t)gm * rs_stride];
        if (residual) v += residual[(size_t)gm * N_ + gn];
        const size_t o = (size_t)gm * N_ + gn;
        if (outF) outF[o] = accumulate ? (outF[o] + v) : v;
        if (outB) outB[o] = (__bf16)v;
      }
    }
  }
}

// ---------------------------------------------------------------- pooling

__global__ __launch_bounds__(256) void pool_kernel(const float* __restrict__ sr,
                                                   float* __restrict__ xp) {
  using namespace cfg;
  const int bp = blockIdx.x;  // b*P + p
  const int seg = N / P;      // 256
  const float* bsrc = sr + (size_t)bp * seg * D;
  for (int d = threadIdx.x; d < D; d += blockDim.x) {
    float s = 0.f;
    for (int t = 0; t < seg; ++t) s += bsrc[(size_t)t * D + d];
    xp[(size_t)bp * D + d] = s / (float)seg;
  }
}

// -------------------------------------------------- q/k prep: l2, softplus, RoPE

__global__ __launch_bounds__(256) void prep_qk_kernel(
    const float* __restrict__ q, const float* __restrict__ kv,
    const float* __restrict__ temp, float* __restrict__ qsn,
    float* __restrict__ kn) {
  using namespace cfg;
  const int widx = blockIdx.x * 8 + (threadIdx.x >> 5);
  const int lane = threadIdx.x & 31;
  if (widx >= B * H * N) return;
  const int n = widx % N;
  const int h = (widx / N) % H;
  const int b = widx / (N * H);

  const float inv = powf(10000.0f, -(float)lane / (float)HALF);
  const float ang = (float)n * inv;
  const float c = cosf(ang), s = sinf(ang);

  // q: l2 -> softplus(temp)*log(K-1+P) -> rope
  const size_t qoff = ((size_t)(b * N + n)) * D + h * HDIM;
  float q1 = q[qoff + lane], q2 = q[qoff + lane + HALF];
  const float ss = wave_sum(q1 * q1 + q2 * q2);
  const float nrm = fmaxf(sqrtf(ss), 1e-12f);
  const float tsc = log1pf(expf(temp[h])) * logf((float)(K - 1 + P));
  const float sc = tsc / nrm;
  q1 *= sc;
  q2 *= sc;
  const size_t oo = (((size_t)(b * H + h)) * N + n) * HDIM;
  qsn[oo + lane] = q1 * c - q2 * s;
  qsn[oo + lane + HALF] = q1 * s + q2 * c;

  // k: rope -> l2
  const size_t koff = ((size_t)(b * N + n)) * (2 * D) + h * HDIM;
  const float k1 = kv[koff + lane], k2 = kv[koff + lane + HALF];
  const float kr1 = k1 * c - k2 * s;
  const float kr2 = k1 * s + k2 * c;
  const float ks = wave_sum(kr1 * kr1 + kr2 * kr2);
  const float knrm = fmaxf(sqrtf(ks), 1e-12f);
  kn[oo + lane] = kr1 / knrm;
  kn[oo + lane + HALF] = kr2 / knrm;
}

// pooled keys: rope only (no l2)
__global__ void prep_kp_kernel(const float* __restrict__ kvp,
                               float* __restrict__ kPn) {
  using namespace cfg;
  const int idx = blockIdx.x * blockDim.x + threadIdx.x;
  if (idx >= B * H * P * HALF) return;
  const int i = idx % HALF;
  const int p = (idx / HALF) % P;
  const int h = (idx / (HALF * P)) % H;
  const int b = idx / (HALF * P * H);
  const float inv = powf(10000.0f, -(float)i / (float)HALF);
  const float ang = (float)p * inv;
  const float c = cosf(ang), s = sinf(ang);
  const size_t src = ((size_t)(b * P + p)) * (2 * D) + h * HDIM;
  const float t1 = kvp[src + i], t2 = kvp[src + i + HALF];
  const size_t dst = (((size_t)(b * H + h)) * P + p) * HDIM;
  kPn[dst + i] = t1 * c - t2 * s;
  kPn[dst + i + HALF] = t1 * s + t2 * c;
}

// -------------------------------------------------- local+pool attention

__global__ __launch_bounds__(256) void attn_kernel(
    const float* __restrict__ qsn, const float* __restrict__ kn,
    const float* __restrict__ kPn, const float* __restrict__ kv,
    const float* __restrict__ kvp, float* __restrict__ outF,
    __bf16* __restrict__ outB) {
  using namespace cfg;
  const int widx = blockIdx.x * 8 + (threadIdx.x >> 5);
  const int lane = threadIdx.x & 31;
  if (widx >= B * H * N) return;
  const int n = widx % N;
  const int h = (widx / N) % H;
  const int b = widx / (N * H);

  const size_t qoff = (((size_t)(b * H + h)) * N + n) * HDIM;
  const float q1 = qsn[qoff + lane], q2 = qsn[qoff + lane + HALF];

  float sc[K + P];
#pragma unroll
  for (int j = 0; j < K; ++j) {
    const int pos = n + j - K / 2;
    const int pc = min(max(pos, 0), N - 1);
    const size_t ko = (((size_t)(b * H + h)) * N + pc) * HDIM;
    const float d = wave_sum(q1 * kn[ko + lane] + q2 * kn[ko + lane + HALF]);
    sc[j] = (pos < 0 || pos >= N) ? -INFINITY : d;
  }
#pragma unroll
  for (int p = 0; p < P; ++p) {
    const size_t ko = (((size_t)(b * H + h)) * P + p) * HDIM;
    sc[K + p] = wave_sum(q1 * kPn[ko + lane] + q2 * kPn[ko + lane + HALF]);
  }

  float mx = sc[0];
#pragma unroll
  for (int t = 1; t < K + P; ++t) mx = fmaxf(mx, sc[t]);
  float wgt[K + P];
  float se = 0.f;
#pragma unroll
  for (int t = 0; t < K + P; ++t) {
    wgt[t] = __expf(sc[t] - mx);
    se += wgt[t];
  }
  const float isum = 1.0f / se;

  float o1 = 0.f, o2 = 0.f;
#pragma unroll
  for (int j = 0; j < K; ++j) {
    const int pos = min(max(n + j - K / 2, 0), N - 1);
    const size_t vo = ((size_t)(b * N + pos)) * (2 * D) + D + h * HDIM;
    const float w = wgt[j] * isum;
    o1 += w * kv[vo + lane];
    o2 += w * kv[vo + lane + HALF];
  }
#pragma unroll
  for (int p = 0; p < P; ++p) {
    const size_t vo = ((size_t)(b * P + p)) * (2 * D) + D + h * HDIM;
    const float w = wgt[K + p] * isum;
    o1 += w * kvp[vo + lane];
    o2 += w * kvp[vo + lane + HALF];
  }
  const size_t oo = ((size_t)(b * N + n)) * D + h * HDIM;
  outF[oo + lane] = o1;
  outF[oo + lane + HALF] = o2;
  outB[oo + lane] = (__bf16)o1;
  outB[oo + lane + HALF] = (__bf16)o2;
}

// ---------------------------------------------------------------- router

__global__ void zero_stats_kernel(float* __restrict__ stats) {
  if (threadIdx.x < 2 * cfg::E) stats[threadIdx.x] = 0.0f;
}

__global__ __launch_bounds__(256) void router_kernel(
    const float* __restrict__ xn2, const float* __restrict__ rw,
    const float* __restrict__ rb, float* __restrict__ gates,
    float* __restrict__ stats) {
  using namespace cfg;
  const int m = blockIdx.x * 8 + (threadIdx.x >> 5);
  const int lane = threadIdx.x & 31;
  if (m >= M) return;
  float acc[E] = {};
  for (int d = lane; d < D; d += 32) {
    const float xv = xn2[(size_t)m * D + d];
#pragma unroll
    for (int e = 0; e < E; ++e) acc[e] += xv * rw[d * E + e];
  }
#pragma unroll
  for (int e = 0; e < E; ++e) acc[e] = wave_sum(acc[e]) + rb[e];
  if (lane == 0) {
    float mx = acc[0];
    for (int e = 1; e < E; ++e) mx = fmaxf(mx, acc[e]);
    float g[E];
    float se = 0.f;
    for (int e = 0; e < E; ++e) {
      g[e] = __expf(acc[e] - mx);
      se += g[e];
    }
    int arg = 0;
    float best = g[0];
    for (int e = 1; e < E; ++e)
      if (g[e] > best) { best = g[e]; arg = e; }
    for (int e = 0; e < E; ++e) {
      const float gv = g[e] / se;
      gates[(size_t)m * E + e] = gv;
      atomicAdd(&stats[E + e], gv);
    }
    atomicAdd(&stats[arg], 1.0f);
  }
}

// ---------------------------------------------------------------- final

__global__ void final_kernel(const float* __restrict__ x1,
                             const float* __restrict__ moe,
                             const float* __restrict__ stats,
                             float* __restrict__ out, int n, float inv_tok) {
  const int i = blockIdx.x * blockDim.x + threadIdx.x;
  if (i < n) out[i] = x1[i] + moe[i];
  if (i == 0) {
    float aux = 0.f;
    for (int e = 0; e < cfg::E; ++e)
      aux += (stats[e] * inv_tok) * (stats[cfg::E + e] * inv_tok);
    out[n] = 0.01f * (float)cfg::E * aux;
  }
}

// ================================================================ launcher

extern "C" void kernel_launch(void* const* d_in, const int* in_sizes, int n_in,
                              void* d_out, int out_size, void* d_ws,
                              size_t ws_size, hipStream_t stream) {
  using namespace cfg;
  (void)in_sizes; (void)n_in; (void)out_size; (void)ws_size;

  const float* x    = (const float*)d_in[0];
  const float* temp = (const float*)d_in[1];
  const float* q_w  = (const float*)d_in[2];
  const float* q_b  = (const float*)d_in[3];
  const float* kv_w = (const float*)d_in[4];
  const float* kv_b = (const float*)d_in[5];
  const float* pj_w = (const float*)d_in[6];
  const float* pj_b = (const float*)d_in[7];
  const float* n1g  = (const float*)d_in[8];
  const float* n1b  = (const float*)d_in[9];
  const float* n2g  = (const float*)d_in[10];
  const float* n2b  = (const float*)d_in[11];
  const float* sr_w = (const float*)d_in[12];
  const float* sr_b = (const float*)d_in[13];
  const float* npg  = (const float*)d_in[14];
  const float* npb  = (const float*)d_in[15];
  const float* rt_w = (const float*)d_in[16];
  const float* rt_b = (const float*)d_in[17];
  const float* ew1  = (const float*)d_in[18];
  const float* eb1  = (const float*)d_in[19];
  const float* ew2  = (const float*)d_in[20];
  const float* eb2  = (const float*)d_in[21];
  const float* sw1  = (const float*)d_in[22];
  const float* sb1  = (const float*)d_in[23];
  const float* sw2  = (const float*)d_in[24];
  const float* sb2  = (const float*)d_in[25];

  char* base = (char*)d_ws;
  size_t off = 0;
  auto alloc = [&](size_t bytes) -> char* {
    char* p = base + off;
    off += (bytes + 255) & ~(size_t)255;
    return p;
  };

  float*  xn_f   = (float*) alloc((size_t)M * D * 4);
  __bf16* xn_bf  = (__bf16*)alloc((size_t)M * D * 2);
  __bf16* wqT    = (__bf16*)alloc((size_t)D * D * 2);       // [D,D]  (N,K)
  __bf16* wkvT   = (__bf16*)alloc((size_t)2 * D * D * 2);   // [2D,D]
  __bf16* wsrT   = (__bf16*)alloc((size_t)D * D * 2);
  __bf16* wpjT   = (__bf16*)alloc((size_t)D * D * 2);
  __bf16* we1T   = (__bf16*)alloc((size_t)E * HID * D * 2); // e:[HID,D]
  __bf16* we2T   = (__bf16*)alloc((size_t)E * D * HID * 2); // e:[D,HID]
  __bf16* ws1T   = (__bf16*)alloc((size_t)HID * D * 2);
  __bf16* ws2T   = (__bf16*)alloc((size_t)D * HID * 2);
  float*  qbuf   = (float*) alloc((size_t)M * D * 4);
  float*  kvbuf  = (float*) alloc((size_t)M * 2 * D * 4);
  float*  srbuf  = (float*) alloc((size_t)M * D * 4);
  float*  qsn    = (float*) alloc((size_t)M * D * 4);
  float*  kn     = (float*) alloc((size_t)M * D * 4);
  float*  xp_raw = (float*) alloc((size_t)B * P * D * 4);
  float*  xp_f   = (float*) alloc((size_t)B * P * D * 4);
  __bf16* xp_bf  = (__bf16*)alloc((size_t)B * P * D * 2);
  float*  kvp    = (float*) alloc((size_t)B * P * 2 * D * 4);
  float*  kPn    = (float*) alloc((size_t)B * H * P * HDIM * 4);
  float*  ao_f   = (float*) alloc((size_t)M * D * 4);
  __bf16* ao_bf  = (__bf16*)alloc((size_t)M * D * 2);
  float*  x1     = (float*) alloc((size_t)M * D * 4);
  float*  xn2_f  = (float*) alloc((size_t)M * D * 4);
  __bf16* xn2_bf = (__bf16*)alloc((size_t)M * D * 2);
  float*  gates  = (float*) alloc((size_t)M * E * 4);
  __bf16* hbuf   = (__bf16*)alloc((size_t)M * HID * 2);
  float*  moe    = (float*) alloc((size_t)M * D * 4);
  float*  stats  = (float*) alloc(64);

  auto convT = [&](const float* s, __bf16* d, int Kd, int Nd) {
    const int tot = Kd * Nd;
    f2bf_t_kernel<<<(tot + 255) / 256, 256, 0, stream>>>(s, d, Kd, Nd);
  };
  convT(q_w, wqT, D, D);
  convT(kv_w, wkvT, D, 2 * D);
  convT(sr_w, wsrT, D, D);
  convT(pj_w, wpjT, D, D);
  for (int e = 0; e < E; ++e) {
    convT(ew1 + (size_t)e * D * HID, we1T + (size_t)e * HID * D, D, HID);
    convT(ew2 + (size_t)e * HID * D, we2T + (size_t)e * D * HID, HID, D);
  }
  convT(sw1, ws1T, D, HID);
  convT(sw2, ws2T, HID, D);

  auto gemm = [&](const __bf16* A, const __bf16* WT, const float* bias,
                  const float* resid, const float* rsc, int rstride, float* oF,
                  __bf16* oB, int Mm, int Nn, int Kk, int act, int accum) {
    dim3 g((Nn + 127) / 128, (Mm + 127) / 128);
    gemm_bf16_wmma<<<g, 256, 0, stream>>>(A, WT, bias, resid, rsc, rstride, oF,
                                          oB, Mm, Nn, Kk, act, accum);
  };

  // ---- attention branch
  ln_kernel<<<M, 256, 0, stream>>>(x, n1g, n1b, 1e-6f, xn_f, xn_bf, D);
  gemm(xn_bf, wqT, q_b, nullptr, nullptr, 0, qbuf, nullptr, M, D, D, 0, 0);
  gemm(xn_bf, wkvT, kv_b, nullptr, nullptr, 0, kvbuf, nullptr, M, 2 * D, D, 0, 0);
  gemm(xn_bf, wsrT, sr_b, nullptr, nullptr, 0, srbuf, nullptr, M, D, D, 1, 0);

  pool_kernel<<<B * P, 256, 0, stream>>>(srbuf, xp_raw);
  ln_kernel<<<B * P, 256, 0, stream>>>(xp_raw, npg, npb, 1e-5f, xp_f, xp_bf, D);
  gemm(xp_bf, wkvT, kv_b, nullptr, nullptr, 0, kvp, nullptr, B * P, 2 * D, D, 0, 0);

  prep_qk_kernel<<<(B * H * N + 7) / 8, 256, 0, stream>>>(qbuf, kvbuf, temp, qsn, kn);
  prep_kp_kernel<<<(B * H * P * HALF + 255) / 256, 256, 0, stream>>>(kvp, kPn);
  attn_kernel<<<(B * H * N + 7) / 8, 256, 0, stream>>>(qsn, kn, kPn, kvbuf, kvp,
                                                       ao_f, ao_bf);
  // x1 = x + attn_out @ proj + b
  gemm(ao_bf, wpjT, pj_b, x, nullptr, 0, x1, nullptr, M, D, D, 0, 0);

  // ---- MoE branch
  ln_kernel<<<M, 256, 0, stream>>>(x1, n2g, n2b, 1e-6f, xn2_f, xn2_bf, D);
  zero_stats_kernel<<<1, 32, 0, stream>>>(stats);
  router_kernel<<<(M + 7) / 8, 256, 0, stream>>>(xn2_f, rt_w, rt_b, gates, stats);

  // shared expert writes moe, then experts accumulate gate-scaled outputs
  gemm(xn2_bf, ws1T, sb1, nullptr, nullptr, 0, nullptr, hbuf, M, HID, D, 1, 0);
  gemm(hbuf, ws2T, sb2, nullptr, nullptr, 0, moe, nullptr, M, D, HID, 0, 0);
  for (int e = 0; e < E; ++e) {
    gemm(xn2_bf, we1T + (size_t)e * HID * D, eb1 + (size_t)e * HID, nullptr,
         nullptr, 0, nullptr, hbuf, M, HID, D, 1, 0);
    gemm(hbuf, we2T + (size_t)e * D * HID, eb2 + (size_t)e * D, nullptr,
         gates + e, E, moe, nullptr, M, D, HID, 0, 1);
  }

  final_kernel<<<(M * D + 255) / 256, 256, 0, stream>>>(
      x1, moe, stats, (float*)d_out, M * D, 1.0f / (float)M);
}
